// KerasMultiLIFLayerSparse_35862976921617
// MI455X (gfx1250) — matrix operand order repaired and checked by reading, not compiled, not run on hardware
//
#include <hip/hip_runtime.h>
#include <hip/hip_bf16.h>

typedef __attribute__((ext_vector_type(16))) _Float16 v16h;
typedef __attribute__((ext_vector_type(8)))  float    v8f;

#define T_STEPS 200
#define BATCH   64

// ---------------------------------------------------------------------------
// Scatter sparse input spike ids into a dense [B, 2048] f32 count matrix.
// One block per batch row; 256 threads (= SPARSE[0]).
// ---------------------------------------------------------------------------
__global__ void scatter_input(const float* __restrict__ ids,   // [B,256] this t
                              const int*   __restrict__ num,   // [B]     this t
                              float* __restrict__ X0)          // [B,2048]
{
    const int b   = blockIdx.x;
    const int tid = threadIdx.x;
    float* row = X0 + b * 2048;
    for (int i = tid; i < 2048; i += 256) row[i] = 0.0f;
    __syncthreads();
    const int nv = num[b];
    if (tid < nv) {
        int idx = (int)ids[b * 256 + tid];
        atomicAdd(&row[idx], 1.0f);
    }
}

// ---------------------------------------------------------------------------
// Fused GEMM (spikes @ W^T) + LIF update via V_WMMA_F32_16X16X32_F16.
// Block = 128 threads = 4 waves; block tile = 64(M=batch) x 32(N=neurons).
// Each wave owns M rows [16w,16w+16) and BOTH 16-wide N tiles (2 accumulators)
// so the A fragment is reused across 2 WMMAs -> 4 wmma / K-stage / wave.
// K staged 64-wide through LDS as f16 (fp32 W converted on the fly; W stays
// L2-resident: 14MB << 192MB). Next K-stage of W is prefetched
// (global_prefetch_b8) while the current stage computes.
// Epilogue: v = decay*S + I ; state_out = v ; S = v - (v>th)*th ; Xout = spike.
// ---------------------------------------------------------------------------
__global__ void lif_gemm(const float* __restrict__ Xin,    // [B, n_in] dense spikes
                         const float* __restrict__ W,      // [n_out, n_in] fp32
                         const float* __restrict__ decay,  // [n_out]
                         const float* __restrict__ thresh, // [n_out]
                         float* __restrict__ S,            // [B, n_out] running state
                         float* __restrict__ state_out,    // [B, n_out] this t
                         float* __restrict__ Xout,         // [B, n_out] dense spikes
                         int n_in, int n_out)
{
    __shared__ _Float16 Xs[BATCH][64 + 8];   // +8 halfs padding vs bank conflicts
    __shared__ _Float16 Ws[32][64 + 8];

    const int tid  = threadIdx.x;
    const int wid  = tid >> 5;
    const int lane = tid & 31;
    const int n0   = blockIdx.x * 32;
    const int m0   = wid * 16;

    const int half_id = lane >> 4;     // 0: lanes 0-15, 1: lanes 16-31
    const int mr      = m0 + (lane & 15);
    const int nc      = lane & 15;

    v8f acc0 = {};
    v8f acc1 = {};

    for (int k0 = 0; k0 < n_in; k0 += 64) {
        // Prefetch next K-stage of the weight tile into cache (global_prefetch_b8)
        if (k0 + 64 < n_in)
            __builtin_prefetch(&W[(size_t)(n0 + (tid & 31)) * n_in + k0 + 64
                                  + ((tid >> 5) << 4)], 0, 3);

        // Stage A tile: 64x64 halfs, 128 threads -> 32 elems each
        for (int i = tid; i < BATCH * 64; i += 128) {
            int r = i >> 6, c = i & 63;
            Xs[r][c] = (_Float16)Xin[r * n_in + k0 + c];
        }
        // Stage B tile (32 neurons x 64 k), converting fp32 -> f16
        for (int i = tid; i < 32 * 64; i += 128) {
            int r = i >> 6, c = i & 63;
            Ws[r][c] = (_Float16)W[(size_t)(n0 + r) * n_in + k0 + c];
        }
        __syncthreads();

#pragma unroll
        for (int ks = 0; ks < 2; ++ks) {
            const int kb = ks * 32;
            v16h a, b0, b1;
            // A 16x32 f16 layout (ISA 7.12.2): lanes 0-15 rows M, VGPR0-3 K=0..7,
            // VGPR4-7 K=16..23; lanes 16-31 shifted by +8 in K.
#pragma unroll
            for (int j = 0; j < 8; ++j) {
                int ka = (j < 4 ? 2 * j : 16 + 2 * (j - 4)) + 8 * half_id;
                a[2 * j]     = Xs[mr][kb + ka];
                a[2 * j + 1] = Xs[mr][kb + ka + 1];
                // B 32x16: same K order as A; VGPR0-3 K=0..7, VGPR4-7 K=16..23,
                // lanes 16-31 shifted by +8 in K, 2 K-rows packed per VGPR.
                int kB = (j < 4 ? 2 * j : 16 + 2 * (j - 4)) + 8 * half_id;
                b0[2 * j]     = Ws[nc][kb + kB];
                b0[2 * j + 1] = Ws[nc][kb + kB + 1];
                b1[2 * j]     = Ws[16 + nc][kb + kB];
                b1[2 * j + 1] = Ws[16 + nc][kb + kB + 1];
            }
            acc0 = __builtin_amdgcn_wmma_f32_16x16x32_f16(
                false, a, false, b0, (short)0, acc0, false, false);
            acc1 = __builtin_amdgcn_wmma_f32_16x16x32_f16(
                false, a, false, b1, (short)0, acc1, false, false);
        }
        __syncthreads();
    }

    // LIF epilogue from C layout: c[r] -> (M = r + 8*half_id, N = lane&15)
#pragma unroll
    for (int t = 0; t < 2; ++t) {
        const int   col = n0 + t * 16 + nc;
        const float dc  = decay[col];
        const float th  = thresh[col];
#pragma unroll
        for (int r = 0; r < 8; ++r) {
            const int brow = m0 + r + 8 * half_id;
            const size_t o = (size_t)brow * n_out + col;
            float v = dc * S[o] + (t == 0 ? acc0[r] : acc1[r]);
            state_out[o] = v;                       // pre-reset membrane potential
            float s = (v > th) ? 1.0f : 0.0f;
            S[o]    = v - s * th;                   // soft reset
            Xout[o] = s;                            // dense spikes for next layer
        }
    }
}

// ---------------------------------------------------------------------------
// Stable-argsort sparsify: one block (256 thr) per batch row of n elements.
// pos(i) = prefix(i) if spike else total + (i - prefix(i)); write if pos < K.
// ---------------------------------------------------------------------------
__global__ void sparsify(const float* __restrict__ Xdense, // [B, n]
                         float* __restrict__ out_ids,      // [B, K] this t
                         int*   __restrict__ out_num,      // [B]    this t
                         int n, int K)
{
    __shared__ int scan[256];
    const int b     = blockIdx.x;
    const int tid   = threadIdx.x;
    const int chunk = n / 256;
    const float* row = Xdense + (size_t)b * n;

    int cnt = 0;
    for (int i = 0; i < chunk; ++i)
        cnt += (row[tid * chunk + i] > 0.5f) ? 1 : 0;
    scan[tid] = cnt;
    __syncthreads();

    for (int off = 1; off < 256; off <<= 1) {
        int v   = scan[tid];
        int add = (tid >= off) ? scan[tid - off] : 0;
        __syncthreads();
        scan[tid] = v + add;
        __syncthreads();
    }

    const int total = scan[255];
    int p = scan[tid] - cnt;            // exclusive prefix for this thread
    float* oid = out_ids + (size_t)b * K;
    for (int i = 0; i < chunk; ++i) {
        const int gi = tid * chunk + i;
        const bool sp = row[gi] > 0.5f;
        const int pos = sp ? p : (total + (gi - p));
        if (sp) ++p;
        if (pos < K) oid[pos] = (float)gi;
    }
    if (tid == 0) out_num[b] = total;
}

// ---------------------------------------------------------------------------
// Host launcher
// ---------------------------------------------------------------------------
extern "C" void kernel_launch(void* const* d_in, const int* in_sizes, int n_in_args,
                              void* d_out, int out_size, void* d_ws, size_t ws_size,
                              hipStream_t stream)
{
    // Input order: inp_spike_ids, num_inp_spikes, then per-layer {w, init, decay, thresh}
    const float* inp_ids = (const float*)d_in[0];   // [T,B,256] f32
    const int*   inp_num = (const int*)  d_in[1];   // [T,B,1]   i32
    const float* w[3]     = {(const float*)d_in[2], (const float*)d_in[6],  (const float*)d_in[10]};
    const float* init_s[3]= {(const float*)d_in[3], (const float*)d_in[7],  (const float*)d_in[11]};
    const float* dec[3]   = {(const float*)d_in[4], (const float*)d_in[8],  (const float*)d_in[12]};
    const float* thr[3]   = {(const float*)d_in[5], (const float*)d_in[9],  (const float*)d_in[13]};

    const int NOUT[3] = {1024, 1024, 512};
    const int NIN [3] = {2048, 1024, 1024};
    const int K  [3] = {128, 128, 64};

    // Workspace carve (floats)
    float* X0 = (float*)d_ws;                 // [64,2048] dense input spikes
    float* X1 = X0 + BATCH * 2048;            // [64,1024] layer0 spikes
    float* X2 = X1 + BATCH * 1024;            // [64,1024] layer1 spikes
    float* X3 = X2 + BATCH * 1024;            // [64,512]  layer2 spikes
    float* S0 = X3 + BATCH * 512;             // running states
    float* S1 = S0 + BATCH * 1024;
    float* S2 = S1 + BATCH * 1024;
    float* Sl[3] = {S0, S1, S2};
    float* Xo[3] = {X1, X2, X3};

    // Initialize running state from inputs (deterministic each call)
    for (int l = 0; l < 3; ++l)
        hipMemcpyAsync(Sl[l], init_s[l], sizeof(float) * BATCH * NOUT[l],
                       hipMemcpyDeviceToDevice, stream);

    // Output carve: (ids0,ids1,ids2, num0,num1,num2, st0,st1,st2)
    float* oid0 = (float*)d_out;
    float* oid1 = oid0 + (size_t)T_STEPS * BATCH * 128;
    float* oid2 = oid1 + (size_t)T_STEPS * BATCH * 128;
    int*   onum0 = (int*)(oid2 + (size_t)T_STEPS * BATCH * 64);
    int*   onum1 = onum0 + (size_t)T_STEPS * BATCH;
    int*   onum2 = onum1 + (size_t)T_STEPS * BATCH;
    float* ost0 = (float*)(onum2 + (size_t)T_STEPS * BATCH);
    float* ost1 = ost0 + (size_t)T_STEPS * BATCH * 1024;
    float* ost2 = ost1 + (size_t)T_STEPS * BATCH * 1024;
    float* oid[3]  = {oid0, oid1, oid2};
    int*   onum[3] = {onum0, onum1, onum2};
    float* ost[3]  = {ost0, ost1, ost2};

    for (int t = 0; t < T_STEPS; ++t) {
        scatter_input<<<BATCH, 256, 0, stream>>>(
            inp_ids + (size_t)t * BATCH * 256, inp_num + (size_t)t * BATCH, X0);

        const float* xin = X0;
        for (int l = 0; l < 3; ++l) {
            lif_gemm<<<NOUT[l] / 32, 128, 0, stream>>>(
                xin, w[l], dec[l], thr[l], Sl[l],
                ost[l] + (size_t)t * BATCH * NOUT[l], Xo[l], NIN[l], NOUT[l]);
            sparsify<<<BATCH, 256, 0, stream>>>(
                Xo[l], oid[l] + (size_t)t * BATCH * K[l],
                onum[l] + (size_t)t * BATCH, NOUT[l], K[l]);
            xin = Xo[l];
        }
    }
}
